// AWQGemm_79422535238420
// MI455X (gfx1250) — matrix operand order repaired and checked by reading, not compiled, not run on hardware
//
#include <hip/hip_runtime.h>
#include <hip/hip_fp16.h>

typedef __attribute__((ext_vector_type(16))) _Float16 v16h;
typedef __attribute__((ext_vector_type(8)))  _Float16 v8h;
typedef __attribute__((ext_vector_type(8)))  float    v8f;
typedef __attribute__((ext_vector_type(4)))  float    v4f;

#define M_DIM 1024
#define K_DIM 4096
#define N_DIM 11008
#define NPACK (N_DIM / 8)      // 1376 packed int32 columns
#define GS    128              // AWQ group size
#define BLK_M 128
#define BLK_N 128
#define BLK_K 32
#define LDH   40               // halves per row/col (32 + 8 pad; multiple of 8 -> 16B-aligned chunks)

// Nibble for unpacked column j lives at bit 4*AWQ_ORDER[j]; AWQ_ORDER=(0,4,1,5,2,6,3,7)
__device__ __constant__ int kAwqShift[8] = {0, 16, 4, 20, 8, 24, 12, 28};

__global__ __launch_bounds__(256)
void awq_wmma_f16_kernel(const float* __restrict__ x,
                         const int*   __restrict__ qweight,
                         const float* __restrict__ scales,
                         const int*   __restrict__ qzeros,
                         float* __restrict__ out)
{
    // Double-buffered tiles: 2 * (10KB + 10KB) = 40KB of the 320KB WGP LDS
    __shared__ _Float16 Alds[2][BLK_M * LDH];   // [row][k], padded
    __shared__ _Float16 Blds[2][BLK_N * LDH];   // [n][k] column-major tile, padded

    const int tid  = threadIdx.x;
    const int lane = tid & 31;
    const int wave = tid >> 5;
    const int wm   = wave >> 1;              // 0..3 : 32-row band
    const int wn   = wave & 1;               // 0..1 : 64-col band

    const int n0 = blockIdx.x * BLK_N;       // output col base
    const int m0 = blockIdx.y * BLK_M;       // output row base

    // A staging: thread -> (row, 8-float chunk), two passes for 128 rows
    const int rowA = tid >> 2;               // 0..63
    const int kcA  = (tid & 3) << 3;         // 0,8,16,24

    // B staging: thread -> (packed col, pair of k rows)
    const int pc = tid & 15;                 // packed col within tile (0..15)
    const int rr = tid >> 4;                 // 0..15 -> k rows 2rr, 2rr+1

    auto stageA = [&](int buf, int k0) {
        #pragma unroll
        for (int s = 0; s < 2; ++s) {
            const int row = rowA + s * 64;
            const float* xp = &x[(size_t)(m0 + row) * K_DIM + k0 + kcA];
            v4f x0 = *(const v4f*)(xp);
            v4f x1 = *(const v4f*)(xp + 4);
            v8h ah;
            #pragma unroll
            for (int i = 0; i < 4; ++i) {
                ah[i]     = (_Float16)x0[i];
                ah[i + 4] = (_Float16)x1[i];
            }
            *(v8h*)&Alds[buf][row * LDH + kcA] = ah;
        }
    };

    auto stageB = [&](int buf, int k0) {
        const int g   = k0 / GS;                         // group (constant within k-step)
        const int pcg = (n0 >> 3) + pc;                  // global packed column
        const int q0  = qweight[(size_t)(k0 + 2 * rr)     * NPACK + pcg];
        const int q1  = qweight[(size_t)(k0 + 2 * rr + 1) * NPACK + pcg];
        const int z   = qzeros[(size_t)g * NPACK + pcg];
        const float* sp = &scales[(size_t)g * N_DIM + n0 + pc * 8];
        v4f s0 = *(const v4f*)(sp);
        v4f s1 = *(const v4f*)(sp + 4);
        #pragma unroll
        for (int j = 0; j < 8; ++j) {
            const int   sh = kAwqShift[j];
            const float sj = (j < 4) ? s0[j] : s1[j - 4];
            const float zf = (float)((z  >> sh) & 0xF);
            const float w0 = ((float)((q0 >> sh) & 0xF) - zf) * sj;
            const float w1 = ((float)((q1 >> sh) & 0xF) - zf) * sj;
            _Float16* bp = &Blds[buf][(pc * 8 + j) * LDH + 2 * rr];
            bp[0] = (_Float16)w0;   // k = 2rr
            bp[1] = (_Float16)w1;   // k = 2rr+1
        }
    };

    v8f c[2][4] = {};

    // Prologue: fill buffer 0
    stageA(0, 0);
    stageB(0, 0);
    __syncthreads();

    for (int k0 = 0; k0 < K_DIM; k0 += BLK_K) {
        const int cur  = (k0 >> 5) & 1;
        const int lrow = lane & 15;
        const int khi  = lane >> 4;   // 0 or 1 (half-wave select)

        // ---------- issue fragment loads from current buffer first (DS in-order) ----------
        v16h a[2], b[4];
        #pragma unroll
        for (int mt = 0; mt < 2; ++mt) {
            // A 16x32 f16: lanes 0-15 hold K 0-7 & 16-23; lanes 16-31 hold K 8-15 & 24-31
            const _Float16* pa = &Alds[cur][(wm * 32 + mt * 16 + lrow) * LDH + (khi << 3)];
            v8h lo = *(const v8h*)(pa);        // K base .. base+7
            v8h hi = *(const v8h*)(pa + 16);   // K base+16 .. base+23
            a[mt] = __builtin_shufflevector(lo, hi, 0,1,2,3,4,5,6,7,8,9,10,11,12,13,14,15);
        }
        #pragma unroll
        for (int nt = 0; nt < 4; ++nt) {
            // B 32x16 f16: col = lane%16; lanes 0-15 hold K 0-7 & 16-23; lanes 16-31 hold K 8-15 & 24-31
            const _Float16* pb = &Blds[cur][(wn * 64 + nt * 16 + lrow) * LDH + (khi << 3)];
            v8h lo = *(const v8h*)(pb);
            v8h hi = *(const v8h*)(pb + 16);
            b[nt] = __builtin_shufflevector(lo, hi, 0,1,2,3,4,5,6,7,8,9,10,11,12,13,14,15);
        }

        // ---------- stage next K-step into the other buffer (overlaps with WMMA) ----------
        if (k0 + BLK_K < K_DIM) {
            stageA(cur ^ 1, k0 + BLK_K);
            stageB(cur ^ 1, k0 + BLK_K);
        }
        if (k0 + 2 * BLK_K < K_DIM) {
            __builtin_prefetch(&qweight[(size_t)(k0 + 2 * BLK_K + 2 * rr) * NPACK + (n0 >> 3) + pc], 0, 0);
            __builtin_prefetch(&x[(size_t)(m0 + rowA) * K_DIM + k0 + 2 * BLK_K + kcA], 0, 0);
        }

        // ---------- 8 WMMAs per wave per K-step ----------
        #pragma unroll
        for (int mt = 0; mt < 2; ++mt)
            #pragma unroll
            for (int nt = 0; nt < 4; ++nt)
                c[mt][nt] = __builtin_amdgcn_wmma_f32_16x16x32_f16(
                    /*neg_a=*/false, a[mt], /*neg_b=*/false, b[nt],
                    /*c_mod=*/(short)0, c[mt][nt],
                    /*reuse_a=*/false, /*reuse_b=*/false);

        __syncthreads();
    }

    // ---------- epilogue: C/D layout -> global (VGPR i: M=i (lanes 0-15), M=i+8 (lanes 16-31)) ----------
    {
        const int lrow = lane & 15;
        const int lhi  = lane >> 4;
        #pragma unroll
        for (int mt = 0; mt < 2; ++mt) {
            #pragma unroll
            for (int nt = 0; nt < 4; ++nt) {
                const int col = n0 + wn * 64 + nt * 16 + lrow;
                #pragma unroll
                for (int i = 0; i < 8; ++i) {
                    const int row = m0 + wm * 32 + mt * 16 + lhi * 8 + i;
                    out[(size_t)row * N_DIM + col] = c[mt][nt][i];
                }
            }
        }
    }
}

extern "C" void kernel_launch(void* const* d_in, const int* in_sizes, int n_in,
                              void* d_out, int out_size, void* d_ws, size_t ws_size,
                              hipStream_t stream) {
    const float* x       = (const float*)d_in[0];
    const int*   qweight = (const int*)  d_in[1];
    const float* scales  = (const float*)d_in[2];
    const int*   qzeros  = (const int*)  d_in[3];
    float*       out     = (float*)d_out;

    dim3 grid(N_DIM / BLK_N, M_DIM / BLK_M);   // 86 x 8
    awq_wmma_f16_kernel<<<grid, 256, 0, stream>>>(x, qweight, scales, qzeros, out);
}
